// PatientGraphTransformer_32392643346796
// MI455X (gfx1250) — matrix operand order repaired and checked by reading, not compiled, not run on hardware
//
#include <hip/hip_runtime.h>
#include <hip/hip_bf16.h>

// ---------------- model constants ----------------
#define NN      4096      // nodes
#define NFEAT   128
#define HID     256
#define NHEAD   4
#define NHE     1024
#define NGRP    32
#define OUTD    128
#define NEDGE   65536
#define NHEI    32768
#define FFD     2048
#define ETOT    (NEDGE + NN)

typedef __attribute__((ext_vector_type(16))) __bf16 bf16x16;
typedef __attribute__((ext_vector_type(8)))  float  f32x8;
typedef __attribute__((ext_vector_type(8)))  unsigned short u16x8;

__device__ __forceinline__ unsigned short f2bfu(float f) {
  unsigned u = __float_as_uint(f);
  u += 0x7FFFu + ((u >> 16) & 1u);        // RNE
  return (unsigned short)(u >> 16);
}
__device__ __forceinline__ __bf16 f2bf(float f) {
  return __builtin_bit_cast(__bf16, f2bfu(f));
}
__device__ __forceinline__ __bf16 u2bf(unsigned short s) {
  return __builtin_bit_cast(__bf16, s);
}

__device__ __forceinline__ void atomicMaxF(float* addr, float val) {
  if (val >= 0.f) atomicMax((int*)addr, __float_as_int(val));
  else            atomicMin((unsigned int*)addr, __float_as_uint(val));
}

// ---- CDNA5 async global->LDS copy (16B per lane), tracked by ASYNCcnt ----
__device__ __forceinline__ void g2l16(const void* g, void* lds) {
  unsigned loff = (unsigned)(unsigned long long)lds;   // LDS aperture: addr[31:0]
  asm volatile("global_load_async_to_lds_b128 %0, %1, off"
               :: "v"(loff), "v"(g) : "memory");
}
__device__ __forceinline__ void async_wait() {
  asm volatile("s_wait_asynccnt 0" ::: "memory");
}

// fp32 -> bf16 bulk convert (n multiple of 8)
__global__ void k_f2bf(const float* __restrict__ in, unsigned short* __restrict__ out, int n) {
  int i = (blockIdx.x * 256 + threadIdx.x) * 8;
  if (i >= n) return;
  float4 a = *(const float4*)(in + i);
  float4 b = *(const float4*)(in + i + 4);
  u16x8 v;
  v[0] = f2bfu(a.x); v[1] = f2bfu(a.y); v[2] = f2bfu(a.z); v[3] = f2bfu(a.w);
  v[4] = f2bfu(b.x); v[5] = f2bfu(b.y); v[6] = f2bfu(b.z); v[7] = f2bfu(b.w);
  *(u16x8*)(out + i) = v;
}

// =====================================================================
// Tiled GEMM:  C = act(Abf[M,K] @ Wbf[N,K]^T + bias), bf16 in, f32/bf16 out
// act: 0 none, 1 relu, 2 sigmoid, 3 qkv (scale cols<256 by 0.125)
// Double-buffered LDS; async copy of tile k+1 overlaps WMMA on tile k.
// =====================================================================
#define BM 128
#define BN 64
#define BK 32
#define LSA 40    // LDS row stride (halves); 80B, 16B aligned

__global__ __launch_bounds__(256) void k_gemm_bf(
    const unsigned short* __restrict__ A, const unsigned short* __restrict__ W,
    const float* __restrict__ bias, float* __restrict__ Cf,
    unsigned short* __restrict__ Cb,
    int M, int N, int K, int ldc, int act)
{
  __shared__ __bf16 As[2][BM * LSA];
  __shared__ __bf16 Ws[2][BN * LSA];
  const int tid  = threadIdx.x;
  const int wid  = tid >> 5;
  const int lane = tid & 31;
  const int half = lane >> 4;
  const int l16  = lane & 15;
  const int mb = blockIdx.x, nb = blockIdx.y;

  f32x8 acc[4];
#pragma unroll
  for (int j = 0; j < 4; ++j)
#pragma unroll
    for (int v = 0; v < 8; ++v) acc[j][v] = 0.f;

  const int arow = tid >> 1;            // 0..127
  const int ac   = (tid & 1) * 16;      // halves 0/16
  const int wrow = tid >> 2;            // 0..63
  const int wc   = (tid & 3) * 8;       // halves
  int ga = mb * BM + arow;
  if (ga >= M) ga = M - 1;              // branchless guard (dup rows discarded)
  const int gw = nb * BN + wrow;
  const unsigned short* Abase = A + (long)ga * K;
  const unsigned short* Wbase = W + (long)gw * K;

  auto stage = [&](int kk, int buf) {
    const unsigned short* as = Abase + kk + ac;
    g2l16(as,     &As[buf][arow * LSA + ac]);
    g2l16(as + 8, &As[buf][arow * LSA + ac + 8]);
    g2l16(Wbase + kk + wc, &Ws[buf][wrow * LSA + wc]);
  };

  stage(0, 0);
  async_wait();
  __syncthreads();

  int cur = 0;
  for (int kk = 0; kk < K; kk += BK) {
    const bool last = (kk + BK >= K);
    if (!last) {
      stage(kk + BK, cur ^ 1);          // overlap next-tile copy with WMMAs
      if (kk + 2 * BK < K) {
        __builtin_prefetch(Abase + kk + 2 * BK + ac, 0, 0);
        __builtin_prefetch(Wbase + kk + 2 * BK + wc, 0, 0);
      }
    }
    const __bf16* ap = &As[cur][(wid * 16 + l16) * LSA];
    bf16x16 afrag;
#pragma unroll
    for (int e = 0; e < 8; ++e) {
      afrag[e]     = ap[half * 8 + e];
      afrag[8 + e] = ap[16 + half * 8 + e];
    }
#pragma unroll
    for (int j = 0; j < 4; ++j) {
      const __bf16* bp = &Ws[cur][(j * 16 + l16) * LSA + half * 8];
      bf16x16 bfrag;
#pragma unroll
      for (int e = 0; e < 16; ++e) bfrag[e] = bp[(e >> 3) * 16 + (e & 7)];
      acc[j] = __builtin_amdgcn_wmma_f32_16x16x32_bf16(
          false, afrag, false, bfrag, (short)0, acc[j], false, false);
    }
    if (!last) {
      async_wait();
      __syncthreads();
    }
    cur ^= 1;
  }

#pragma unroll
  for (int j = 0; j < 4; ++j) {
    const int col = nb * BN + j * 16 + l16;
    const float b = bias ? bias[col] : 0.f;
#pragma unroll
    for (int v = 0; v < 8; ++v) {
      const int row = mb * BM + wid * 16 + v + half * 8;
      if (row < M) {
        float val = acc[j][v] + b;
        if (act == 1)      val = fmaxf(val, 0.f);
        else if (act == 2) val = 1.f / (1.f + __expf(-val));
        else if (act == 3) { if (col < 256) val *= 0.125f; }  // pre-scale Q
        if (Cb) Cb[(long)row * ldc + col] = f2bfu(val);
        else    Cf[(long)row * ldc + col] = val;
      }
    }
  }
}

// =====================================================================
// Flash attention: qkv bf16 rows [q|k|v] (768 halves), out bf16 [L,256]
// grid (L/64, H), block 128 (4 waves); double-buffered K/V tiles.
// =====================================================================
#define FB 64
#define FS 80

__global__ __launch_bounds__(128) void k_flash(
    const unsigned short* __restrict__ qkv, unsigned short* __restrict__ out, int L)
{
  __shared__ __bf16 Qs[FB * FS];
  __shared__ __bf16 Ps[FB * FS];
  __shared__ __bf16 Ks[2][FB * FS];
  __shared__ __bf16 Vt[2][FB * FS];
  const int tid  = threadIdx.x;
  const int wid  = tid >> 5;
  const int lane = tid & 31;
  const int half = lane >> 4;
  const int l16  = lane & 15;
  const int h    = blockIdx.y;
  const int row0 = blockIdx.x * FB;

  auto stageKV = [&](int kb, int buf) {
    for (int c = tid; c < 512; c += 128) {            // K rows, async
      int r = c >> 3, co = (c & 7) * 8;
      g2l16(qkv + (long)(kb * FB + r) * 768 + 256 + h * 64 + co, &Ks[buf][r * FS + co]);
    }
    for (int idx = tid; idx < FB * 64; idx += 128) {  // V transposed, scalar
      int r = idx >> 6, cc = idx & 63;
      Vt[buf][cc * FS + r] = u2bf(qkv[(long)(kb * FB + r) * 768 + 512 + h * 64 + cc]);
    }
  };

  // Q block (pre-scaled by 0.125 in the QKV gemm) + first K/V tile
  for (int c = tid; c < 512; c += 128) {
    int r = c >> 3, co = (c & 7) * 8;
    g2l16(qkv + (long)(row0 + r) * 768 + h * 64 + co, &Qs[r * FS + co]);
  }
  stageKV(0, 0);
  async_wait();
  __syncthreads();

  const __bf16* qp = &Qs[(wid * 16 + l16) * FS];
  bf16x16 qa0, qa1;
#pragma unroll
  for (int e = 0; e < 8; ++e) {
    qa0[e]     = qp[half * 8 + e];
    qa0[8 + e] = qp[16 + half * 8 + e];
    qa1[e]     = qp[32 + half * 8 + e];
    qa1[8 + e] = qp[48 + half * 8 + e];
  }

  float mrow[8], lrow[8];
  f32x8 oacc[4];
#pragma unroll
  for (int v = 0; v < 8; ++v) { mrow[v] = -3.0e38f; lrow[v] = 0.f; }
#pragma unroll
  for (int j = 0; j < 4; ++j)
#pragma unroll
    for (int v = 0; v < 8; ++v) oacc[j][v] = 0.f;

  const int nkb = L / FB;
  int cur = 0;
  for (int kb = 0; kb < nkb; ++kb) {
    if (kb + 1 < nkb) stageKV(kb + 1, cur ^ 1);   // overlap with compute

    f32x8 s[4];
#pragma unroll
    for (int j = 0; j < 4; ++j) {
      const __bf16* kp = &Ks[cur][(j * 16 + l16) * FS + half * 8];
      bf16x16 b0, b1;
#pragma unroll
      for (int e = 0; e < 16; ++e) { b0[e] = kp[(e >> 3) * 16 + (e & 7)]; b1[e] = kp[32 + (e >> 3) * 16 + (e & 7)]; }
      f32x8 sj;
#pragma unroll
      for (int v = 0; v < 8; ++v) sj[v] = 0.f;
      sj = __builtin_amdgcn_wmma_f32_16x16x32_bf16(false, qa0, false, b0, (short)0, sj, false, false);
      sj = __builtin_amdgcn_wmma_f32_16x16x32_bf16(false, qa1, false, b1, (short)0, sj, false, false);
      s[j] = sj;
    }

    // online softmax; row r = v + half*8 lives in 16 lanes of a half-wave
#pragma unroll
    for (int v = 0; v < 8; ++v) {
      float mloc = fmaxf(fmaxf(s[0][v], s[1][v]), fmaxf(s[2][v], s[3][v]));
#pragma unroll
      for (int off = 8; off > 0; off >>= 1) mloc = fmaxf(mloc, __shfl_xor(mloc, off, 32));
      float mnew = fmaxf(mrow[v], mloc);
      float sc   = __expf(mrow[v] - mnew);
      float ps   = 0.f;
#pragma unroll
      for (int j = 0; j < 4; ++j) {
        float p = __expf(s[j][v] - mnew);
        s[j][v] = p;
        ps += p;
      }
#pragma unroll
      for (int off = 8; off > 0; off >>= 1) ps += __shfl_xor(ps, off, 32);
      lrow[v] = lrow[v] * sc + ps;
      mrow[v] = mnew;
#pragma unroll
      for (int j = 0; j < 4; ++j) oacc[j][v] *= sc;
      const int pr = wid * 16 + v + half * 8;
#pragma unroll
      for (int j = 0; j < 4; ++j) Ps[pr * FS + j * 16 + l16] = f2bf(s[j][v]);
    }

    const __bf16* pp = &Ps[(wid * 16 + l16) * FS];
    bf16x16 pa0, pa1;
#pragma unroll
    for (int e = 0; e < 8; ++e) {
      pa0[e]     = pp[half * 8 + e];
      pa0[8 + e] = pp[16 + half * 8 + e];
      pa1[e]     = pp[32 + half * 8 + e];
      pa1[8 + e] = pp[48 + half * 8 + e];
    }
#pragma unroll
    for (int j = 0; j < 4; ++j) {
      const __bf16* vp = &Vt[cur][(j * 16 + l16) * FS + half * 8];
      bf16x16 vb0, vb1;
#pragma unroll
      for (int e = 0; e < 16; ++e) { vb0[e] = vp[(e >> 3) * 16 + (e & 7)]; vb1[e] = vp[32 + (e >> 3) * 16 + (e & 7)]; }
      oacc[j] = __builtin_amdgcn_wmma_f32_16x16x32_bf16(false, pa0, false, vb0, (short)0, oacc[j], false, false);
      oacc[j] = __builtin_amdgcn_wmma_f32_16x16x32_bf16(false, pa1, false, vb1, (short)0, oacc[j], false, false);
    }

    if (kb + 1 < nkb) {
      async_wait();
      __syncthreads();
    }
    cur ^= 1;
  }

#pragma unroll
  for (int j = 0; j < 4; ++j)
#pragma unroll
    for (int v = 0; v < 8; ++v) {
      const int r = row0 + wid * 16 + v + half * 8;
      out[(long)r * 256 + h * 64 + j * 16 + l16] = f2bfu(oacc[j][v] / lrow[v]);
    }
}

// =====================================================================
// elementwise / segment kernels
// =====================================================================
__global__ void k_fill(float* p, float v, int n) {
  int i = blockIdx.x * 256 + threadIdx.x;
  if (i < n) p[i] = v;
}
__global__ void k_add(const float* a, const float* b, float* c, int n) {
  int i = blockIdx.x * 256 + threadIdx.x;
  if (i < n) c[i] = a[i] + b[i];
}

__global__ __launch_bounds__(256) void k_add_ln(
    const float* __restrict__ a, const float* __restrict__ b,
    const float* __restrict__ g, const float* __restrict__ be,
    float* __restrict__ out, int n)
{
  const int wid = threadIdx.x >> 5, lane = threadIdx.x & 31;
  const int r = blockIdx.x * 8 + wid;
  if (r >= n) return;
  float vals[8];
  float s = 0.f;
#pragma unroll
  for (int k = 0; k < 8; ++k) {
    int d = lane + k * 32;
    vals[k] = a[(long)r * 256 + d] + b[(long)r * 256 + d];
    s += vals[k];
  }
#pragma unroll
  for (int off = 16; off > 0; off >>= 1) s += __shfl_xor(s, off, 32);
  const float mu = s * (1.f / 256.f);
  float var = 0.f;
#pragma unroll
  for (int k = 0; k < 8; ++k) { float d = vals[k] - mu; var += d * d; }
#pragma unroll
  for (int off = 16; off > 0; off >>= 1) var += __shfl_xor(var, off, 32);
  const float inv = rsqrtf(var * (1.f / 256.f) + 1e-5f);
#pragma unroll
  for (int k = 0; k < 8; ++k) {
    int d = lane + k * 32;
    out[(long)r * 256 + d] = (vals[k] - mu) * inv * g[d] + be[d];
  }
}

// ---- hypergraph ----
__global__ void k_hyper_count(const int* node, const int* edge, float* cnt, float* deg) {
  int i = blockIdx.x * 256 + threadIdx.x;
  if (i >= NHEI) return;
  atomicAdd(&cnt[edge[i]], 1.f);
  atomicAdd(&deg[node[i]], 1.f);
}
__global__ void k_hyper_hattr(const int* node, const int* edge, const float* x, float* hattr) {
  int t = blockIdx.x * 256 + threadIdx.x;
  if (t >= NHEI * 32) return;
  int i = t >> 5, c = (t & 31) * 4;
  int nd = node[i], e = edge[i];
#pragma unroll
  for (int d = 0; d < 4; ++d)
    atomicAdd(&hattr[(long)e * NFEAT + c + d], x[(long)nd * NFEAT + c + d]);
}
__global__ void k_div_rows(float* m, const float* cnt, int rows, int cols) {
  int t = blockIdx.x * 256 + threadIdx.x;
  if (t >= rows * cols) return;
  m[t] /= fmaxf(cnt[t / cols], 1.f);
}
__global__ void k_head_dot(const float* __restrict__ xt, const float* __restrict__ att,
                           int att_stride, int att_off, float* out, int n) {
  int t = blockIdx.x * 256 + threadIdx.x;
  if (t >= n * 4) return;
  int i = t >> 2, h = t & 3;
  const float* xr = xt + (long)i * 1024 + h * 256;
  const float* ar = att + h * att_stride + att_off;
  float s = 0.f;
  for (int d = 0; d < 256; ++d) s += xr[d] * ar[d];
  out[t] = s;
}
__global__ void k_hyper_alpha(const int* node, const int* edge, const float* sx,
                              const float* se, float* alpha, float* segm) {
  int t = blockIdx.x * 256 + threadIdx.x;
  if (t >= NHEI * 4) return;
  int i = t >> 2, h = t & 3;
  float a = sx[node[i] * 4 + h] + se[edge[i] * 4 + h];
  a = a > 0.f ? a : 0.2f * a;
  alpha[t] = a;
  atomicMaxF(&segm[node[i] * 4 + h], a);
}
__global__ void k_expsum(const float* segm, const int* seg, float* alpha, float* segs, int M) {
  int t = blockIdx.x * 256 + threadIdx.x;
  if (t >= M * 4) return;
  int i = t >> 2, h = t & 3;
  float m = segm[seg[i] * 4 + h];
  if (m < -3.0e37f) m = 0.f;
  float e = __expf(alpha[t] - m);
  alpha[t] = e;
  atomicAdd(&segs[seg[i] * 4 + h], e);
}
__global__ void k_norm(const int* seg, float* alpha, const float* segs, int M) {
  int t = blockIdx.x * 256 + threadIdx.x;
  if (t >= M * 4) return;
  int i = t >> 2, h = t & 3;
  alpha[t] /= (segs[seg[i] * 4 + h] + 1e-16f);
}
__global__ void k_hyper_m(const int* node, const int* edge, const float* alpha,
                          const float* cnt, const float* xt, float* mbuf) {
  int t = blockIdx.x * 256 + threadIdx.x;
  if (t >= NHEI * 256) return;
  int i = t >> 8, h = (t >> 6) & 3, c = (t & 63) * 4;
  int nd = node[i], e = edge[i];
  float cc = cnt[e];
  float coef = (cc > 0.f ? 1.f / cc : 0.f) * alpha[i * 4 + h];
  long bx = (long)nd * 1024 + h * 256 + c;
  long bm = (long)e * 1024 + h * 256 + c;
#pragma unroll
  for (int d = 0; d < 4; ++d) atomicAdd(&mbuf[bm + d], coef * xt[bx + d]);
}
__global__ void k_hyper_out(const int* node, const int* edge, const float* alpha,
                            const float* deg, const float* mbuf, float* outacc) {
  int t = blockIdx.x * 256 + threadIdx.x;
  if (t >= NHEI * 256) return;
  int i = t >> 8, h = (t >> 6) & 3, c = (t & 63) * 4;
  int nd = node[i], e = edge[i];
  float dd = deg[nd];
  float coef = (dd > 0.f ? 1.f / dd : 0.f) * alpha[i * 4 + h];
  long bm = (long)e * 1024 + h * 256 + c;
  long bo = (long)nd * 1024 + h * 256 + c;
#pragma unroll
  for (int d = 0; d < 4; ++d) atomicAdd(&outacc[bo + d], coef * mbuf[bm + d]);
}
__global__ void k_headmean(const float* acc, const float* bias, float* out, int n, int relu) {
  int t = blockIdx.x * 256 + threadIdx.x;
  if (t >= n * 256) return;
  int r = t >> 8, d = t & 255;
  float s = 0.f;
#pragma unroll
  for (int h = 0; h < 4; ++h) s += acc[(long)r * 1024 + h * 256 + d];
  s = s * 0.25f + bias[d];
  if (relu) s = fmaxf(s, 0.f);
  out[t] = s;
}

// ---- GAT ----
__global__ void k_make_edges(const int* ei, int* src, int* dst) {
  int i = blockIdx.x * 256 + threadIdx.x;
  if (i >= ETOT) return;
  if (i < NEDGE) { src[i] = ei[i]; dst[i] = ei[NEDGE + i]; }
  else           { src[i] = i - NEDGE; dst[i] = i - NEDGE; }
}
__global__ void k_edge_alpha(const int* src, const int* dst, const float* ss,
                             const float* sd, float* alpha, float* segm) {
  int t = blockIdx.x * 256 + threadIdx.x;
  if (t >= ETOT * 4) return;
  int i = t >> 2, h = t & 3;
  float a = ss[src[i] * 4 + h] + sd[dst[i] * 4 + h];
  a = a > 0.f ? a : 0.2f * a;
  alpha[t] = a;
  atomicMaxF(&segm[dst[i] * 4 + h], a);
}
__global__ void k_gat_out(const int* src, const int* dst, const float* alpha,
                          const float* xt, float* outacc) {
  int t = blockIdx.x * 256 + threadIdx.x;
  if (t >= ETOT * 256) return;
  int i = t >> 8, h = (t >> 6) & 3, c = (t & 63) * 4;
  float coef = alpha[i * 4 + h];
  long bx = (long)src[i] * 1024 + h * 256 + c;
  long bo = (long)dst[i] * 1024 + h * 256 + c;
#pragma unroll
  for (int d = 0; d < 4; ++d) atomicAdd(&outacc[bo + d], coef * xt[bx + d]);
}

// ---- pooling ----
__global__ void k_pool_acc(const int* batch, const float* h, float* pooled, float* gcnt) {
  int t = blockIdx.x * 256 + threadIdx.x;
  if (t >= NN * 256) return;
  int r = t >> 8, d = t & 255;
  int g = batch[r];
  atomicAdd(&pooled[(long)g * 256 + d], h[t]);
  if (d == 0) atomicAdd(&gcnt[g], 1.f);
}
__global__ void k_pool_div(float* pooled, const float* gcnt) {
  int t = blockIdx.x * 256 + threadIdx.x;
  if (t >= NGRP * 256) return;
  pooled[t] /= fmaxf(gcnt[t >> 8], 1.f);
}

// =====================================================================
// host side
// =====================================================================
struct Net {
  const float *x, *meta;
  const int *ei, *hei, *batch;
  const float *hy_W, *hy_att, *hy_bias;
  const float *gat_W[2], *gat_asrc[2], *gat_adst[2], *gat_bias[2];
  const float *tx_Win[2], *tx_bin[2], *tx_Wout[2], *tx_bout[2];
  const float *tx_W1[2], *tx_b1f[2], *tx_W2[2], *tx_b2f[2];
  const float *tx_g1[2], *tx_be1[2], *tx_g2[2], *tx_be2[2];
  const float *v_Win[3], *v_bin[3], *v_Wout[3], *v_bout[3];
  const float *mv_W, *mv_b, *fc_W, *fc_b, *meta_W, *meta_b, *link_W, *link_b;
};

static void resolve(void* const* din, const int* sz, Net& P) {
  auto F = [&](int i) { return (const float*)din[i]; };
  auto I = [&](int i) { return (const int*)din[i]; };
  bool topSorted = (sz[0] == 4096);
  int pb;
  bool paramsSorted;
  if (topSorted) {
    P.batch = I(0); P.ei = I(1); P.hei = I(2); P.meta = F(3); P.x = F(59);
    pb = 4; paramsSorted = true;
  } else {
    P.x = F(0); P.ei = I(1); P.hei = I(2); P.batch = I(3); P.meta = F(4);
    pb = 5;
    paramsSorted = (sz[5] != 131072);
  }
  if (paramsSorted) {
    P.fc_W = F(pb + 0); P.fc_b = F(pb + 1);
    for (int l = 0; l < 2; ++l) {
      int b = pb + 2 + l * 4;
      P.gat_W[l] = F(b); P.gat_adst[l] = F(b + 1); P.gat_asrc[l] = F(b + 2); P.gat_bias[l] = F(b + 3);
    }
    P.hy_W = F(pb + 10); P.hy_att = F(pb + 11); P.hy_bias = F(pb + 12);
    P.link_W = F(pb + 13); P.link_b = F(pb + 14);
    P.meta_W = F(pb + 15); P.meta_b = F(pb + 16);
    P.mv_W = F(pb + 17); P.mv_b = F(pb + 18);
    for (int l = 0; l < 2; ++l) {
      int b = pb + 19 + l * 12;
      P.tx_W1[l] = F(b); P.tx_W2[l] = F(b + 1); P.tx_Win[l] = F(b + 2); P.tx_Wout[l] = F(b + 3);
      P.tx_b1f[l] = F(b + 4); P.tx_b2f[l] = F(b + 5); P.tx_be1[l] = F(b + 6); P.tx_be2[l] = F(b + 7);
      P.tx_bin[l] = F(b + 8); P.tx_bout[l] = F(b + 9); P.tx_g1[l] = F(b + 10); P.tx_g2[l] = F(b + 11);
    }
    for (int v = 0; v < 3; ++v) {
      int b = pb + 43 + v * 4;
      P.v_Win[v] = F(b); P.v_Wout[v] = F(b + 1); P.v_bin[v] = F(b + 2); P.v_bout[v] = F(b + 3);
    }
  } else {
    P.hy_W = F(pb + 0); P.hy_att = F(pb + 1); P.hy_bias = F(pb + 2);
    for (int l = 0; l < 2; ++l) {
      int b = pb + 3 + l * 4;
      P.gat_W[l] = F(b); P.gat_asrc[l] = F(b + 1); P.gat_adst[l] = F(b + 2); P.gat_bias[l] = F(b + 3);
    }
    for (int l = 0; l < 2; ++l) {
      int b = pb + 11 + l * 12;
      P.tx_Win[l] = F(b); P.tx_bin[l] = F(b + 1); P.tx_Wout[l] = F(b + 2); P.tx_bout[l] = F(b + 3);
      P.tx_W1[l] = F(b + 4); P.tx_b1f[l] = F(b + 5); P.tx_W2[l] = F(b + 6); P.tx_b2f[l] = F(b + 7);
      P.tx_g1[l] = F(b + 8); P.tx_be1[l] = F(b + 9); P.tx_g2[l] = F(b + 10); P.tx_be2[l] = F(b + 11);
    }
    for (int v = 0; v < 3; ++v) {
      int b = pb + 35 + v * 4;
      P.v_Win[v] = F(b); P.v_bin[v] = F(b + 1); P.v_Wout[v] = F(b + 2); P.v_bout[v] = F(b + 3);
    }
    P.mv_W = F(pb + 47); P.mv_b = F(pb + 48);
    P.fc_W = F(pb + 49); P.fc_b = F(pb + 50);
    P.meta_W = F(pb + 51); P.meta_b = F(pb + 52);
    P.link_W = F(pb + 53); P.link_b = F(pb + 54);
  }
}

static inline int gs(long n) { return (int)((n + 255) / 256); }

extern "C" void kernel_launch(void* const* d_in, const int* in_sizes, int n_in,
                              void* d_out, int out_size, void* d_ws, size_t ws_size,
                              hipStream_t stream) {
  Net P;
  resolve(d_in, in_sizes, P);

  // ---- workspace layout ----
  float* ws = (float*)d_ws;
  float* H0   = ws;                           // 4096*256
  float* H1   = H0 + (1 << 20);
  float* H2   = H1 + (1 << 20);
  float* XT   = H2 + (1 << 20);               // 4096*1024 (per-head features, f32)
  float* BIGB = XT + (long)4096 * 1024;       // 4096*1024 (et / outacc / vcat)
  float* BIGC = BIGB + (long)4096 * 1024;     // 1M floats (hyper mbuf)
  float* SM   = BIGC + (1 << 20);             // stats region (1M floats)
  float* cnt    = SM;
  float* deg    = SM + 1024;
  float* segm   = SM + 8192;
  float* segs   = SM + 24576;
  float* sx     = SM + 40960;
  float* se     = SM + 57344;
  float* hattr  = SM + 81920;
  float* alphaH = SM + 212992;
  float* alphaE = SM + 344064;
  int*   srcA   = (int*)(SM + 622592);
  int*   dstA   = (int*)(SM + 692224);
  float* pooled = SM + 761856;
  float* gcnt   = SM + 770048;
  // bf16 arenas
  unsigned short* WBF   = (unsigned short*)(SM + (1 << 20));  // 5M halves (weights)
  unsigned short* BFQ   = WBF + (5L << 20);                   // 8M halves (qkv / ff hidden / mv out)
  unsigned short* BFX   = BFQ + (8L << 20);                   // 4M halves (transient A operands)
  unsigned short* BFATT = BFX + (4L << 20);                   // 1M halves (flash out)

  auto cvt = [&](const float* f, unsigned short* b, long n) {
    k_f2bf<<<gs(n / 8), 256, 0, stream>>>(f, b, (int)n);
  };
  auto gemm = [&](const unsigned short* A, const unsigned short* W, const float* bias,
                  float* Cf, unsigned short* Cb, int M, int N, int K, int ldc, int act) {
    dim3 g((M + BM - 1) / BM, N / BN);
    k_gemm_bf<<<g, 256, 0, stream>>>(A, W, bias, Cf, Cb, M, N, K, ldc, act);
  };

  // ---- convert all weights to bf16 once ----
  unsigned short* wb = WBF;
  auto wcvt = [&](const float* f, long n) {
    unsigned short* p = wb; wb += n; cvt(f, p, n); return (const unsigned short*)p;
  };
  const unsigned short* hyWb = wcvt(P.hy_W, 1024L * 128);
  const unsigned short* gatWb[2] = { wcvt(P.gat_W[0], 1024L * 256), wcvt(P.gat_W[1], 1024L * 256) };
  const unsigned short *txWinb[2], *txWoutb[2], *txW1b[2], *txW2b[2];
  for (int l = 0; l < 2; ++l) {
    txWinb[l]  = wcvt(P.tx_Win[l],  768L * 256);
    txWoutb[l] = wcvt(P.tx_Wout[l], 256L * 256);
    txW1b[l]   = wcvt(P.tx_W1[l],   2048L * 256);
    txW2b[l]   = wcvt(P.tx_W2[l],   256L * 2048);
  }
  const unsigned short *vWinb[3], *vWoutb[3];
  for (int v = 0; v < 3; ++v) {
    vWinb[v]  = wcvt(P.v_Win[v],  768L * 256);
    vWoutb[v] = wcvt(P.v_Wout[v], 256L * 256);
  }
  const unsigned short* mvWb   = wcvt(P.mv_W,   256L * 768);
  const unsigned short* fcWb   = wcvt(P.fc_W,   256L * 256);
  const unsigned short* metaWb = wcvt(P.meta_W, 256L * 256);
  const unsigned short* linkWb = wcvt(P.link_W, 128L * 256);

  const int* he_node = P.hei;
  const int* he_edge = P.hei + NHEI;
  const float NEGINF = -__builtin_huge_valf();

  // ============ 1. hypergraph conv ============
  hipMemsetAsync(cnt, 0, (1024 + 4096) * sizeof(float), stream);
  hipMemsetAsync(hattr, 0, (long)NHE * NFEAT * sizeof(float), stream);
  hipMemsetAsync(segs, 0, 16384 * sizeof(float), stream);
  k_fill<<<gs(16384), 256, 0, stream>>>(segm, NEGINF, 16384);
  k_hyper_count<<<gs(NHEI), 256, 0, stream>>>(he_node, he_edge, cnt, deg);
  k_hyper_hattr<<<gs((long)NHEI * 32), 256, 0, stream>>>(he_node, he_edge, P.x, hattr);
  k_div_rows<<<gs((long)NHE * NFEAT), 256, 0, stream>>>(hattr, cnt, NHE, NFEAT);
  cvt(P.x, BFX, (long)NN * NFEAT);
  gemm(BFX, hyWb, nullptr, XT, nullptr, NN, 1024, NFEAT, 1024, 0);       // xt
  cvt(hattr, BFX, (long)NHE * NFEAT);
  gemm(BFX, hyWb, nullptr, BIGB, nullptr, NHE, 1024, NFEAT, 1024, 0);    // et
  k_head_dot<<<gs(NN * 4), 256, 0, stream>>>(XT, P.hy_att, 512, 0, sx, NN);
  k_head_dot<<<gs(NHE * 4), 256, 0, stream>>>(BIGB, P.hy_att, 512, 256, se, NHE);
  k_hyper_alpha<<<gs(NHEI * 4), 256, 0, stream>>>(he_node, he_edge, sx, se, alphaH, segm);
  k_expsum<<<gs(NHEI * 4), 256, 0, stream>>>(segm, he_node, alphaH, segs, NHEI);
  k_norm<<<gs(NHEI * 4), 256, 0, stream>>>(he_node, alphaH, segs, NHEI);
  hipMemsetAsync(BIGC, 0, (long)NHE * 1024 * sizeof(float), stream);
  k_hyper_m<<<gs((long)NHEI * 256), 256, 0, stream>>>(he_node, he_edge, alphaH, cnt, XT, BIGC);
  hipMemsetAsync(BIGB, 0, (long)NN * 1024 * sizeof(float), stream);
  k_hyper_out<<<gs((long)NHEI * 256), 256, 0, stream>>>(he_node, he_edge, alphaH, deg, BIGC, BIGB);
  k_headmean<<<gs((long)NN * 256), 256, 0, stream>>>(BIGB, P.hy_bias, H0, NN, 1);

  // ============ 2. GAT x2 ============
  k_make_edges<<<gs(ETOT), 256, 0, stream>>>(P.ei, srcA, dstA);
  const float* gin = H0;
  float* gout = H1;
  for (int l = 0; l < 2; ++l) {
    cvt(gin, BFX, (long)NN * HID);
    gemm(BFX, gatWb[l], nullptr, XT, nullptr, NN, 1024, HID, 1024, 0);
    k_head_dot<<<gs(NN * 4), 256, 0, stream>>>(XT, P.gat_asrc[l], 256, 0, sx, NN);
    k_head_dot<<<gs(NN * 4), 256, 0, stream>>>(XT, P.gat_adst[l], 256, 0, se, NN);
    k_fill<<<gs(16384), 256, 0, stream>>>(segm, NEGINF, 16384);
    hipMemsetAsync(segs, 0, 16384 * sizeof(float), stream);
    k_edge_alpha<<<gs(ETOT * 4), 256, 0, stream>>>(srcA, dstA, sx, se, alphaE, segm);
    k_expsum<<<gs(ETOT * 4), 256, 0, stream>>>(segm, dstA, alphaE, segs, ETOT);
    k_norm<<<gs(ETOT * 4), 256, 0, stream>>>(dstA, alphaE, segs, ETOT);
    hipMemsetAsync(BIGB, 0, (long)NN * 1024 * sizeof(float), stream);
    k_gat_out<<<gs((long)ETOT * 256), 256, 0, stream>>>(srcA, dstA, alphaE, XT, BIGB);
    k_headmean<<<gs((long)NN * 256), 256, 0, stream>>>(BIGB, P.gat_bias[l], gout, NN, 0);
    const float* t = gin; gin = gout; gout = (float*)t;
  }
  // result back in H0

  // ============ 3. transformer layers x2 ============
  for (int l = 0; l < 2; ++l) {
    cvt(H0, BFX, (long)NN * HID);
    gemm(BFX, txWinb[l], P.tx_bin[l], nullptr, BFQ, NN, 768, HID, 768, 3);   // qkv (Q scaled)
    k_flash<<<dim3(NN / FB, NHEAD), 128, 0, stream>>>(BFQ, BFATT, NN);
    gemm(BFATT, txWoutb[l], P.tx_bout[l], H2, nullptr, NN, HID, HID, HID, 0);
    k_add_ln<<<NN / 8, 256, 0, stream>>>(H0, H2, P.tx_g1[l], P.tx_be1[l], H1, NN);
    cvt(H1, BFX, (long)NN * HID);
    gemm(BFX, txW1b[l], P.tx_b1f[l], nullptr, BFQ, NN, FFD, HID, FFD, 1);    // ff1+relu -> bf16
    gemm(BFQ, txW2b[l], P.tx_b2f[l], H2, nullptr, NN, HID, FFD, HID, 0);
    k_add_ln<<<NN / 8, 256, 0, stream>>>(H1, H2, P.tx_g2[l], P.tx_be2[l], H0, NN);
  }

  // ============ 4. multiview MHAs -> vcat (BIGB, ld 768) ============
  cvt(H0, BFX, (long)NN * HID);
  for (int v = 0; v < 3; ++v) {
    gemm(BFX, vWinb[v], P.v_bin[v], nullptr, BFQ, NN, 768, HID, 768, 3);
    k_flash<<<dim3(NN / FB, NHEAD), 128, 0, stream>>>(BFQ, BFATT, NN);
    gemm(BFATT, vWoutb[v], P.v_bout[v], BIGB + v * 256, nullptr, NN, HID, HID, 768, 0);
  }

  // ============ 5. head ============
  cvt(BIGB, BFX, (long)NN * 768);                                          // vcat
  gemm(BFX, mvWb, P.mv_b, nullptr, BFQ, NN, HID, 768, HID, 1);             // mv -> bf16
  gemm(BFQ, fcWb, P.fc_b, H2, nullptr, NN, HID, HID, HID, 1);              // fc
  cvt(P.meta, BFX, (long)NN * HID);
  gemm(BFX, metaWb, P.meta_b, H0, nullptr, NN, HID, HID, HID, 1);          // meta
  k_add<<<gs((long)NN * 256), 256, 0, stream>>>(H2, H0, H1, NN * 256);

  hipMemsetAsync(pooled, 0, (NGRP * 256 + 64) * sizeof(float), stream);
  k_pool_acc<<<gs((long)NN * 256), 256, 0, stream>>>(P.batch, H1, pooled, gcnt);
  k_pool_div<<<gs(NGRP * 256), 256, 0, stream>>>(pooled, gcnt);
  cvt(pooled, BFX, NGRP * 256);
  gemm(BFX, linkWb, P.link_b, (float*)d_out, nullptr, NGRP, OUTD, HID, OUTD, 2);
  (void)n_in; (void)out_size; (void)ws_size;
}
